// MixtureLinearAttention_72078141161992
// MI455X (gfx1250) — matrix operand order repaired and checked by reading, not compiled, not run on hardware
//
#include <hip/hip_runtime.h>
#include <cstdint>

// ---------------------------------------------------------------------------
// Mixture Linear Attention for MI455X (gfx1250, wave32, WMMA f16 16x16x32).
//
// Dims: N=4, S=2048, H=16, D=64 (feature dim E=64, value dim M=64), C=4.
// ws layout (f32): KV[N][C][H][m=64][e=64]  (4 MB)  then Ksum[N][C][H][64].
//
// Roofline: ~17 GFLOP of dense GEMM vs ~66 MB HBM traffic (2.9 us @23.3TB/s).
// F16 WMMA (16384 FLOP/inst, f32 accumulate) keeps compute near the memory
// time; all operands are O(1) so f16 with f32 accumulation is numerically safe.
// ---------------------------------------------------------------------------

typedef __attribute__((ext_vector_type(16))) _Float16 v16h;
typedef __attribute__((ext_vector_type(8)))  float    v8f;

#define WMMA16(a, b, c) \
  __builtin_amdgcn_wmma_f32_16x16x32_f16(false, (a), false, (b), (short)0, (c), false, false)

constexpr int Nb = 4, Sq = 2048, Hh = 16, Dd = 64, Cc = 4;
constexpr int SPLIT = 8;                 // s-chunks per (n,c,h) for occupancy
constexpr int SCHUNK = Sq / SPLIT;       // 256
constexpr int STEPS  = SCHUNK / 32;      // 8 (each step consumes 32 s-rows)
constexpr float EPS = 1e-6f;

union V16HU {
  v16h     h;
  float4   f4[2];
  uint32_t u[8];
};

__device__ inline v8f v8f_zero() {
  v8f r;
#pragma unroll
  for (int j = 0; j < 8; ++j) r[j] = 0.f;
  return r;
}

// f32 p[0..7], p[16..23] -> v16h (B fragments from global; p 16B aligned)
__device__ inline v16h cvt16(const float* p) {
  float4 f0 = *(const float4*)(p);
  float4 f1 = *(const float4*)(p + 4);
  float4 f2 = *(const float4*)(p + 16);
  float4 f3 = *(const float4*)(p + 20);
  v16h r;
  r[0]=(_Float16)f0.x;  r[1]=(_Float16)f0.y;  r[2]=(_Float16)f0.z;  r[3]=(_Float16)f0.w;
  r[4]=(_Float16)f1.x;  r[5]=(_Float16)f1.y;  r[6]=(_Float16)f1.z;  r[7]=(_Float16)f1.w;
  r[8]=(_Float16)f2.x;  r[9]=(_Float16)f2.y;  r[10]=(_Float16)f2.z; r[11]=(_Float16)f2.w;
  r[12]=(_Float16)f3.x; r[13]=(_Float16)f3.y; r[14]=(_Float16)f3.z; r[15]=(_Float16)f3.w;
  return r;
}

// LDS f16: halves 0..7 from p[0..7], halves 8..15 from p[16..23]
// (A-fragment pattern: K pairs split across lane halves). p 16B aligned.
__device__ inline v16h lds_frag_8_8(const _Float16* p) {
  V16HU x;
  x.f4[0] = *(const float4*)(p);
  x.f4[1] = *(const float4*)(p + 16);
  return x.h;
}

__device__ inline float elu1(float x) { return x > 0.f ? x + 1.f : __expf(x); }

// ---------------------------------------------------------------------------
// Kernel 1: per (n,c,h,s-chunk) accumulate KV[m][e] (64x64) and Ksum[e].
// 4 waves: wave w computes projection e-tile w from LDS-staged f16 tiles,
// shares packed A-fragments via LDS, then owns m-tile w of the state.
// ---------------------------------------------------------------------------
__global__ __launch_bounds__(128) void mla_state_kernel(
    const float* __restrict__ keys, const float* __restrict__ values,
    const float* __restrict__ key_mask, const float* __restrict__ featW,
    float* __restrict__ wsKV, float* __restrict__ wsKsum) {
  __shared__ __align__(32) _Float16 sWT[64 * 64];   // W^T[e][d] f16
  __shared__ __align__(32) _Float16 sKT[32 * 64];   // keys tile [s_local][d]
  __shared__ __align__(32) _Float16 sVT[64 * 32];   // V^T[m][s_local] f16
  __shared__ float sMask[32];
  __shared__ __align__(32) uint32_t sAfrag[4][32][8];  // packed A per e-tile

  const int bid = blockIdx.x;
  const int chunk = bid & (SPLIT - 1);
  const int h = (bid >> 3) & (Hh - 1);
  const int c = (bid >> 7) & (Cc - 1);
  const int n = bid >> 9;
  const int tid = threadIdx.x;
  const int w = tid >> 5, lane = tid & 31;
  const int l16 = lane & 15;
  const int hi8 = (lane & 16) ? 8 : 0;
  const int hi16 = (lane & 16) ? 16 : 0;

  // Stage W^T[e][d] = featW[c][d][e] as f16 (coalesced over e).
  for (int i = tid; i < 64 * 64; i += 128) {
    int d = i >> 6, e = i & 63;
    sWT[e * 64 + d] = (_Float16)featW[(c * 64 + d) * 64 + e];
  }
  __syncthreads();

  // Projection B fragments (K=d, N=e): row e = 16w+l16, halves 8/8 split.
  v16h bW0 = lds_frag_8_8(sWT + (16 * w + l16) * 64 + 0  + hi8);
  v16h bW1 = lds_frag_8_8(sWT + (16 * w + l16) * 64 + 32 + hi8);

  v8f acc[4];  // state C-tiles: (e-tile t, m-tile w), M=e, N=m
#pragma unroll
  for (int t = 0; t < 4; ++t) acc[t] = v8f_zero();
  float ks = 0.f;

  const size_t rowStride = (size_t)Hh * Dd;  // f32 elems between s rows
  const int s0base = chunk * SCHUNK;
  for (int st = 0; st < STEPS; ++st) {
    const int s0 = s0base + st * 32;
    __syncthreads();  // previous step's consumers of sKT/sVT are done
    // Stage keys tile [s][d] and V^T[m][s] (f16) + mask for these 32 rows.
    for (int i = tid; i < 32 * 64; i += 128) {
      int sl = i >> 6, m = i & 63;
      const float* vsrc = &values[(((size_t)n * Sq + s0 + sl) * Hh + h) * Dd + m];
      sVT[m * 32 + sl] = (_Float16)*vsrc;
      const float* ksrc = &keys[(((size_t)n * Sq + s0 + sl) * Hh + h) * Dd + m];
      sKT[sl * 64 + m] = (_Float16)*ksrc;
      if (st + 1 < STEPS && (m & 31) == 0) {   // one prefetch per 128B line
        __builtin_prefetch(vsrc + 32 * rowStride, 0, 3);
        __builtin_prefetch(ksrc + 32 * rowStride, 0, 3);
      }
    }
    if (tid < 32) sMask[tid] = key_mask[(size_t)n * Sq + s0 + tid];
    __syncthreads();

    // Projection for e-tile w, two 16-row s-subtiles: Kphi[s][e].
    v8f cP[2];
#pragma unroll
    for (int ss = 0; ss < 2; ++ss) {
      v8f cc = v8f_zero();
      const _Float16* rowk = sKT + (ss * 16 + l16) * 64;  // A: M=s, K=d
      cc = WMMA16(lds_frag_8_8(rowk + 0 + hi8), bW0, cc);
      cc = WMMA16(lds_frag_8_8(rowk + 32 + hi8), bW1, cc);
#pragma unroll
      for (int j = 0; j < 8; ++j) {
        float ph = elu1(cc[j]) * sMask[ss * 16 + j + hi8];
        cc[j] = ph;
        ks += ph;  // lane's e = 16w + l16
      }
      cP[ss] = cc;
    }
    // Pack: C-tiles (M=s,N=e) -> A-fragment (M=e, K=s in 0..31). Share in LDS.
    V16HU af;
#pragma unroll
    for (int j = 0; j < 8; ++j) {
      af.h[j] = (_Float16)cP[0][j];
      af.h[8 + j] = (_Float16)cP[1][j];
    }
    *(float4*)&sAfrag[w][lane][0] = af.f4[0];
    *(float4*)&sAfrag[w][lane][4] = af.f4[1];
    __syncthreads();

    // KV accumulation: B (K=s, N=m) from V^T rows; m-tile w.
    v16h bv = lds_frag_8_8(sVT + (16 * w + l16) * 32 + hi8);
#pragma unroll
    for (int t = 0; t < 4; ++t) {
      v16h a = *(const v16h*)&sAfrag[t][lane][0];
      acc[t] = WMMA16(a, bv, acc[t]);
    }
  }

  // Reduce partial state into workspace: KV[m][e] and Ksum[e].
  float* KVb = wsKV + (((size_t)n * Cc + c) * Hh + h) * 4096;
  const int m = 16 * w + l16;
#pragma unroll
  for (int t = 0; t < 4; ++t)
#pragma unroll
    for (int v = 0; v < 8; ++v)
      atomicAdd(&KVb[m * 64 + 16 * t + v + hi8], acc[t][v]);
  atomicAdd(&wsKsum[(((size_t)n * Cc + c) * Hh + h) * 64 + 16 * w + l16], ks);
}

// ---------------------------------------------------------------------------
// Kernel 2: per (n, 16-row l-tile, h) compute out[16 x 64].
// Wave w: computes Qphi^T e-tile w (C layout (M=e,N=l) -> packed A (M=l,K=e)),
// shares fragments + Z partials in LDS, then owns output m-tile w.
// ---------------------------------------------------------------------------
__global__ __launch_bounds__(128) void mla_out_kernel(
    const float* __restrict__ queries, const float* __restrict__ featW,
    const float* __restrict__ mixw, const float* __restrict__ wsKV,
    const float* __restrict__ wsKsum, float* __restrict__ out) {
  __shared__ __align__(32) _Float16 sWT4[Cc][64 * 64];  // W^T[e][d] per c
  __shared__ __align__(32) _Float16 sQT[16 * 64];       // queries tile [l][d]
  __shared__ float sKs[Cc][64];
  __shared__ __align__(32) uint32_t sQfrag[4][32][4];   // packed Qphi^T halves
  __shared__ float sZ[4][16];

  const int bid = blockIdx.x;
  const int h = bid & (Hh - 1);
  const int lt = (bid >> 4) & (Sq / 16 - 1);
  const int n = bid >> 11;
  const int tid = threadIdx.x;
  const int w = tid >> 5, lane = tid & 31;
  const int l16 = lane & 15;
  const int hi8 = (lane & 16) ? 8 : 0;
  const int hi16 = (lane & 16) ? 16 : 0;
  const int l0 = lt * 16;

  for (int i = tid; i < Cc * 64 * 64; i += 128) {
    int cc = i >> 12, r = i & 4095, d = r >> 6, e = r & 63;
    sWT4[cc][e * 64 + d] = (_Float16)featW[i];  // featW flat = (c*64+d)*64+e
  }
  for (int i = tid; i < 16 * 64; i += 128) {
    int l = i >> 6, d = i & 63;
    sQT[l * 64 + d] =
        (_Float16)queries[(((size_t)n * Sq + l0 + l) * Hh + h) * Dd + d];
  }
  for (int i = tid; i < Cc * 64; i += 128)
    sKs[i >> 6][i & 63] =
        wsKsum[(((size_t)n * Cc + (i >> 6)) * Hh + h) * 64 + (i & 63)];
  __syncthreads();

  // Softmax over the 4 mix weights (redundant per thread; trivial cost).
  float mw[Cc];
  float mx = -1e30f;
#pragma unroll
  for (int q = 0; q < Cc; ++q) { mw[q] = mixw[q]; mx = fmaxf(mx, mw[q]); }
  float se = 0.f;
#pragma unroll
  for (int q = 0; q < Cc; ++q) { mw[q] = __expf(mw[q] - mx); se += mw[q]; }
  const float ise = 1.f / se;
#pragma unroll
  for (int q = 0; q < Cc; ++q) mw[q] *= ise;

  // Queries B fragments (K=d, N=l), c-independent, from LDS (shared by waves).
  v16h bq0 = lds_frag_8_8(sQT + l16 * 64 + 0 + hi8);
  v16h bq1 = lds_frag_8_8(sQT + l16 * 64 + 32 + hi8);

  float oacc[8];
#pragma unroll
  for (int v = 0; v < 8; ++v) oacc[v] = 0.f;

  for (int c = 0; c < Cc; ++c) {
    __syncthreads();  // fragment/Z buffers free from previous c
    // Qphi^T e-tile w: A = W^T (M=e, K=d), B = queries^T (K=d, N=l).
    v8f cc = v8f_zero();
    const _Float16* wrow = &sWT4[c][(16 * w + l16) * 64];
    cc = WMMA16(lds_frag_8_8(wrow + 0 + hi8), bq0, cc);
    cc = WMMA16(lds_frag_8_8(wrow + 32 + hi8), bq1, cc);
    float z = 0.f;
#pragma unroll
    for (int j = 0; j < 8; ++j) {
      cc[j] = elu1(cc[j]);
      z += cc[j] * sKs[c][16 * w + j + hi8];  // Z partial over this e-tile
    }
    z += __shfl_xor(z, 16, 32);  // combine lane halves -> full e-tile sum
    V16HU pk;
#pragma unroll
    for (int j = 0; j < 8; ++j) pk.h[j] = (_Float16)cc[j];
    *(float4*)&sQfrag[w][lane][0] = pk.f4[0];
    if (lane < 16) sZ[w][lane] = z;
    __syncthreads();

    const float zt = sZ[0][l16] + sZ[1][l16] + sZ[2][l16] + sZ[3][l16];
    const float zinv = mw[c] / (zt + EPS);  // fold mix weight into 1/Z

    // Assemble A fragments (M=l, K=e): tiles {0,1} and {2,3}.
    V16HU a0u, a1u;
    a0u.f4[0] = *(const float4*)&sQfrag[0][lane][0];
    a0u.f4[1] = *(const float4*)&sQfrag[1][lane][0];
    a1u.f4[0] = *(const float4*)&sQfrag[2][lane][0];
    a1u.f4[1] = *(const float4*)&sQfrag[3][lane][0];
    // B from KV[m][e] rows (K=e, N=m), m-tile w.
    const float* kvrow = wsKV + (((size_t)n * Cc + c) * Hh + h) * 4096 +
                         (size_t)(16 * w + l16) * 64;
    v16h b0 = cvt16(kvrow + 0 + hi8);
    v16h b1 = cvt16(kvrow + 32 + hi8);
    v8f nm = v8f_zero();
    nm = WMMA16(a0u.h, b0, nm);
    nm = WMMA16(a1u.h, b1, nm);
    // out tile element (v,lane): l = v + hi8, m = 16w + l16.
#pragma unroll
    for (int v = 0; v < 8; ++v)
      oacc[v] += __shfl(zinv, v + hi8, 32) * nm[v];
  }

#pragma unroll
  for (int v = 0; v < 8; ++v)
    out[(((size_t)n * Sq + l0 + v + hi8) * Hh + h) * Dd + 16 * w + l16] =
        oacc[v];
}

// ---------------------------------------------------------------------------
extern "C" void kernel_launch(void* const* d_in, const int* in_sizes, int n_in,
                              void* d_out, int out_size, void* d_ws,
                              size_t ws_size, hipStream_t stream) {
  (void)in_sizes; (void)n_in; (void)out_size; (void)ws_size;
  const float* queries = (const float*)d_in[0];
  const float* keys = (const float*)d_in[1];
  const float* values = (const float*)d_in[2];
  const float* key_mask = (const float*)d_in[3];
  const float* featW = (const float*)d_in[4];
  const float* mixw = (const float*)d_in[5];
  float* outp = (float*)d_out;

  float* wsKV = (float*)d_ws;  // [N][C][H][64][64]
  const size_t kvElems = (size_t)Nb * Cc * Hh * 64 * 64;
  float* wsKsum = wsKV + kvElems;  // [N][C][H][64]
  const size_t ksElems = (size_t)Nb * Cc * Hh * 64;

  hipMemsetAsync(d_ws, 0, (kvElems + ksElems) * sizeof(float), stream);
  mla_state_kernel<<<Nb * Cc * Hh * SPLIT, 128, 0, stream>>>(
      keys, values, key_mask, featW, wsKV, wsKsum);
  mla_out_kernel<<<Nb * (Sq / 16) * Hh, 128, 0, stream>>>(
      queries, featW, mixw, wsKV, wsKsum, outp);
}
